// TransformerBlock_5875515261295
// MI455X (gfx1250) — matrix operand order repaired and checked by reading, not compiled, not run on hardware
//
#include <hip/hip_runtime.h>
#include <stdint.h>
#include <stddef.h>

// ---------------------------------------------------------------------------
// Types for CDNA5 WMMA (wave32, 16x16x32 bf16 -> f32)
// ---------------------------------------------------------------------------
typedef __attribute__((ext_vector_type(8)))  float  f32x8;
typedef __attribute__((ext_vector_type(8)))  __bf16 bf16x8;
typedef __attribute__((ext_vector_type(16))) __bf16 bf16x16;

static __device__ inline f32x8 zero8() {
    f32x8 z;
#pragma unroll
    for (int i = 0; i < 8; ++i) z[i] = 0.0f;
    return z;
}

// fp32 -> bf16 RNE (bit-level; used in converters, epilogues and the attention P tile)
static __device__ inline __bf16 f2bf(float f) {
    union { float f; uint32_t u; } v; v.f = f;
    uint32_t u = v.u;
    u += 0x7FFFu + ((u >> 16) & 1u);
    union { uint16_t s; __bf16 b; } o; o.s = (uint16_t)(u >> 16);
    return o.b;
}

// Assemble a v16 bf16 fragment from two contiguous 8-element (16B) runs.
static __device__ inline bf16x16 ld8x2(const __bf16* p0, const __bf16* p1) {
    bf16x8 lo = *(const bf16x8*)p0;
    bf16x8 hi = *(const bf16x8*)p1;
    bf16x16 r;
#pragma unroll
    for (int i = 0; i < 8; ++i) { r[i] = lo[i]; r[i + 8] = hi[i]; }
    return r;
}

static __device__ inline f32x8 wmma_bf16(bf16x16 a, bf16x16 b, f32x8 c) {
    return __builtin_amdgcn_wmma_f32_16x16x32_bf16(
        false, a, false, b, (short)0, c, false, false);
}

// ---------------------------------------------------------------------------
// CDNA5 async global->LDS copy (ASYNCcnt-tracked), 16 bytes per lane.
//   LDS address comes from the vdst VGPR; generic->LDS: low 32 bits are the offset.
// ---------------------------------------------------------------------------
static __device__ inline void async_cp16(void* lds_ptr, const void* gptr) {
    uint32_t lds_off = (uint32_t)(uintptr_t)lds_ptr;
    asm volatile("global_load_async_to_lds_b128 %0, %1, off"
                 :: "v"(lds_off), "v"(gptr) : "memory");
}
static __device__ inline void wait_async0() {
    asm volatile("s_wait_asynccnt 0x0" ::: "memory");
}

// ---------------------------------------------------------------------------
// One-time converters
// ---------------------------------------------------------------------------
__global__ __launch_bounds__(256)
void cvt_bf16_kernel(const float* __restrict__ in, __bf16* __restrict__ out, int n)
{
    const int i = (blockIdx.x * 256 + threadIdx.x) * 4;
    if (i < n) {
        const float4 f = *(const float4*)(in + i);
        out[i + 0] = f2bf(f.x); out[i + 1] = f2bf(f.y);
        out[i + 2] = f2bf(f.z); out[i + 3] = f2bf(f.w);
    }
}

// W[K,N] f32 -> Wt[N,K] bf16 (32x32 LDS-tiled transpose)
__global__ __launch_bounds__(256)
void transpose_bf16_kernel(const float* __restrict__ W, __bf16* __restrict__ Wt,
                           int K, int N)
{
    __shared__ float tile[32][33];
    const int tx = threadIdx.x & 31, ty = threadIdx.x >> 5;
    const int n0 = blockIdx.x * 32, k0 = blockIdx.y * 32;
#pragma unroll
    for (int j = 0; j < 4; ++j)
        tile[ty + j * 8][tx] = W[(size_t)(k0 + ty + j * 8) * N + n0 + tx];
    __syncthreads();
#pragma unroll
    for (int j = 0; j < 4; ++j)
        Wt[(size_t)(n0 + ty + j * 8) * K + k0 + tx] = f2bf(tile[tx][ty + j * 8]);
}

// ---------------------------------------------------------------------------
// bf16 GEMM: C[M,N] = act(A[M,K] @ Wt[N,K]^T + bias[N])
// A, Wt both row-major along K (bf16). Block 128x128x64, 256 thr (8 waves),
// wave tile 64x32, 16 WMMA per staged slab. Double-buffered LDS filled by
// async global->LDS copies (ASYNCcnt) overlapped with WMMA compute.
// ---------------------------------------------------------------------------
#define BM 128
#define BN 128
#define BK 64
#define KPAD 72   // 72 elems = 144B row stride (16B multiple)

__global__ __launch_bounds__(256)
void gemm_bf16_kernel(const __bf16* __restrict__ A, const __bf16* __restrict__ Wt,
                      const float* __restrict__ bias,
                      float* __restrict__ Cf, __bf16* __restrict__ Cb,
                      int M, int N, int Kd, int relu)
{
    __shared__ __align__(16) __bf16 As[2][BM][KPAD];
    __shared__ __align__(16) __bf16 Bs[2][BN][KPAD];

    const int tid  = threadIdx.x;
    const int bn   = blockIdx.x, bm = blockIdx.y;
    const int row0 = bm * BM, col0 = bn * BN;
    const int wv   = tid >> 5;
    const int lane = tid & 31;
    const int grp  = lane >> 4;
    const int ln   = lane & 15;
    const int wm   = wv & 1;
    const int wn   = wv >> 1;

    // stage loader: thread covers one 64B half-row (4 async chunks) of A and B
    const int srow = tid >> 1;          // 0..127
    const int sch  = (tid & 1) * 4;     // chunk base (chunks of 8 bf16 = 16B)

    f32x8 acc[4][2];
#pragma unroll
    for (int rt = 0; rt < 4; ++rt)
#pragma unroll
        for (int ct = 0; ct < 2; ++ct) acc[rt][ct] = zero8();

    const int nk = Kd / BK;

    auto stage = [&](int buf, int t) {
        const int k0 = t * BK;
        const __bf16* ga = A  + (size_t)(row0 + srow) * Kd + k0;
        const __bf16* gb = Wt + (size_t)(col0 + srow) * Kd + k0;
#pragma unroll
        for (int i = 0; i < 4; ++i) {
            async_cp16(&As[buf][srow][(sch + i) * 8], ga + (sch + i) * 8);
            async_cp16(&Bs[buf][srow][(sch + i) * 8], gb + (sch + i) * 8);
        }
    };

    stage(0, 0);

    for (int t = 0; t < nk; ++t) {
        const int cur = t & 1;
        wait_async0();          // my copies into `cur` have landed
        __syncthreads();        // everyone's copies into `cur` have landed
        if (t + 1 < nk) stage(cur ^ 1, t + 1);   // overlap next slab with compute

#pragma unroll
        for (int kk = 0; kk < 2; ++kk) {         // two k=32 WMMA steps per slab
            bf16x16 af[4], bfm[2];
#pragma unroll
            for (int rt = 0; rt < 4; ++rt) {
                const __bf16* p = &As[cur][wm * 64 + rt * 16 + ln][kk * 32];
                af[rt] = ld8x2(p + grp * 8, p + 16 + grp * 8);
            }
#pragma unroll
            for (int ct = 0; ct < 2; ++ct) {
                const __bf16* p = &Bs[cur][wn * 32 + ct * 16 + ln][kk * 32 + grp * 8];
                bfm[ct] = ld8x2(p, p + 16);
            }
#pragma unroll
            for (int rt = 0; rt < 4; ++rt)
#pragma unroll
                for (int ct = 0; ct < 2; ++ct)
                    acc[rt][ct] = wmma_bf16(af[rt], bfm[ct], acc[rt][ct]);
        }

        __syncthreads();        // all reads of `cur` done before it is restaged
    }

    // epilogue: bias (+ReLU); emit f32 and/or bf16
#pragma unroll
    for (int rt = 0; rt < 4; ++rt) {
#pragma unroll
        for (int ct = 0; ct < 2; ++ct) {
            const int rowb = row0 + wm * 64 + rt * 16 + grp * 8;
            const int col  = col0 + wn * 32 + ct * 16 + ln;
            const float bv = bias[col];
#pragma unroll
            for (int r = 0; r < 8; ++r) {
                float v = acc[rt][ct][r] + bv;
                if (relu) v = fmaxf(v, 0.0f);
                if (Cf) Cf[(size_t)(rowb + r) * N + col] = v;
                if (Cb) Cb[(size_t)(rowb + r) * N + col] = f2bf(v);
            }
        }
    }
}

// ---------------------------------------------------------------------------
// Causal flash attention, dk = 64, all-bf16 operands.
// Grid: (B*H, T/64); block = 128 (4 waves); wave = one 16-row Q tile.
// Q/K fragments load straight from global (16B contiguous per lane);
// V is LDS-staged for the transpose.
// ---------------------------------------------------------------------------
__global__ __launch_bounds__(128)
void attn_kernel(const __bf16* __restrict__ Q, const __bf16* __restrict__ K,
                 const __bf16* __restrict__ V, __bf16* __restrict__ O,
                 int B, int T, int D, int H)
{
    __shared__ __align__(16) __bf16 Vt[64][40];      // [d][key]
    __shared__ __align__(16) __bf16 Ps[4][16][40];   // per-wave P relayout

    const int bh = blockIdx.x;
    const int b  = bh / H, h = bh % H;
    const int qt = blockIdx.y;
    const int tid  = threadIdx.x;
    const int wv   = tid >> 5;
    const int lane = tid & 31;
    const int grp  = lane >> 4;
    const int ln   = lane & 15;
    const int qbase = qt * 64 + wv * 16;
    const float SC = 0.125f;                         // 1/sqrt(64), applied to S

    // Q fragments (16 rows x 64 dk, bf16 direct from global)
    bf16x16 aQ[2];
    {
        const __bf16* qp = Q + (size_t)(b * T + qbase + ln) * D + h * 64;
#pragma unroll
        for (int s = 0; s < 2; ++s)
            aQ[s] = ld8x2(qp + s * 32 + grp * 8, qp + s * 32 + grp * 8 + 16);
    }

    f32x8 Oacc[4];
#pragma unroll
    for (int ct = 0; ct < 4; ++ct) Oacc[ct] = zero8();
    float m[8], l[8];
#pragma unroll
    for (int r = 0; r < 8; ++r) { m[r] = -1.0e30f; l[r] = 0.0f; }

    const int nchunk = qt * 2 + 2;
    const int lkey = tid >> 2, lseg = tid & 3;       // V stager: key, 16-d segment

    for (int c = 0; c < nchunk; ++c) {
        const int c0 = c * 32;
        __syncthreads();
        // ---- stage V chunk transposed: Vt[d][key] ----
        {
            const __bf16* vp = V + (size_t)(b * T + c0 + lkey) * D + h * 64 + lseg * 16;
            const bf16x8 v0 = *(const bf16x8*)vp;
            const bf16x8 v1 = *(const bf16x8*)(vp + 8);
#pragma unroll
            for (int i = 0; i < 8; ++i) {
                Vt[lseg * 16 + i][lkey]     = v0[i];
                Vt[lseg * 16 + 8 + i][lkey] = v1[i];
            }
        }
        __syncthreads();

        if (c0 <= qbase + 15) {                      // wave-uniform causal guard
            // ---- S = Q K^T, two 16-key tiles, K fragments from global ----
            f32x8 S0 = zero8(), S1 = zero8();
            const __bf16* kbase = K + (size_t)(b * T + c0 + ln) * D + h * 64;
#pragma unroll
            for (int s = 0; s < 2; ++s) {
                const __bf16* k0p = kbase + s * 32 + grp * 8;
                const __bf16* k1p = k0p + (size_t)16 * D;
                S0 = wmma_bf16(aQ[s], ld8x2(k0p, k0p + 16), S0);
                S1 = wmma_bf16(aQ[s], ld8x2(k1p, k1p + 16), S1);
            }
            // ---- causal mask + online softmax ----
            const int k0g = c0 + ln, k1g = c0 + 16 + ln;
#pragma unroll
            for (int r = 0; r < 8; ++r) {
                const int qg = qbase + grp * 8 + r;
                float s0 = (k0g > qg) ? -1.0e30f : S0[r] * SC;
                float s1 = (k1g > qg) ? -1.0e30f : S1[r] * SC;
                float mx = fmaxf(s0, s1);
#pragma unroll
                for (int off = 1; off < 16; off <<= 1)
                    mx = fmaxf(mx, __shfl_xor(mx, off, 16));
                const float mn = fmaxf(m[r], mx);
                const float alpha = __expf(m[r] - mn);
                m[r] = mn;
                const float p0 = __expf(s0 - mn);
                const float p1 = __expf(s1 - mn);
                float rs = p0 + p1;
#pragma unroll
                for (int off = 1; off < 16; off <<= 1)
                    rs += __shfl_xor(rs, off, 16);
                l[r] = l[r] * alpha + rs;
#pragma unroll
                for (int ct = 0; ct < 4; ++ct) Oacc[ct][r] *= alpha;
                Ps[wv][grp * 8 + r][ln]      = f2bf(p0);
                Ps[wv][grp * 8 + r][16 + ln] = f2bf(p1);
            }
            // ---- O += P V ----
            const __bf16* pp = &Ps[wv][ln][0];
            const bf16x16 aP = ld8x2(pp + grp * 8, pp + 16 + grp * 8);
#pragma unroll
            for (int ct = 0; ct < 4; ++ct) {
                const __bf16* vp2 = &Vt[ct * 16 + ln][grp * 8];
                Oacc[ct] = wmma_bf16(aP, ld8x2(vp2, vp2 + 16), Oacc[ct]);
            }
        }
    }

    float inv[8];
#pragma unroll
    for (int r = 0; r < 8; ++r) inv[r] = 1.0f / l[r];
#pragma unroll
    for (int ct = 0; ct < 4; ++ct)
#pragma unroll
        for (int r = 0; r < 8; ++r) {
            const size_t row = (size_t)(b * T + qbase + grp * 8 + r);
            O[row * D + h * 64 + ct * 16 + ln] = f2bf(Oacc[ct][r] * inv[r]);
        }
}

// ---------------------------------------------------------------------------
// out = LayerNorm(x + res)*g + b ; optional bf16 copy (D = 1024, 256 thr/row)
// ---------------------------------------------------------------------------
__global__ __launch_bounds__(256)
void ln_kernel(const float* __restrict__ x, const float* __restrict__ res,
               const float* __restrict__ g, const float* __restrict__ bta,
               float* __restrict__ out, __bf16* __restrict__ outb, int D)
{
    const int row = blockIdx.x, tid = threadIdx.x;
    const float* xr = x   + (size_t)row * D;
    const float* rr = res + (size_t)row * D;
    __shared__ float red[256];

    float v[4]; float s = 0.0f;
#pragma unroll
    for (int j = 0; j < 4; ++j) { v[j] = xr[tid + j * 256] + rr[tid + j * 256]; s += v[j]; }
    red[tid] = s; __syncthreads();
    for (int o = 128; o > 0; o >>= 1) { if (tid < o) red[tid] += red[tid + o]; __syncthreads(); }
    const float mu = red[0] / (float)D;
    __syncthreads();

    float s2 = 0.0f;
#pragma unroll
    for (int j = 0; j < 4; ++j) { const float d = v[j] - mu; s2 += d * d; }
    red[tid] = s2; __syncthreads();
    for (int o = 128; o > 0; o >>= 1) { if (tid < o) red[tid] += red[tid + o]; __syncthreads(); }
    const float rinv = rsqrtf(red[0] / (float)D + 1e-5f);

#pragma unroll
    for (int j = 0; j < 4; ++j) {
        const int i = tid + j * 256;
        const float y = (v[j] - mu) * rinv * g[i] + bta[i];
        out[(size_t)row * D + i] = y;
        if (outb) outb[(size_t)row * D + i] = f2bf(y);
    }
}

// ---------------------------------------------------------------------------
// Orchestration
// ---------------------------------------------------------------------------
extern "C" void kernel_launch(void* const* d_in, const int* in_sizes, int n_in,
                              void* d_out, int out_size, void* d_ws, size_t ws_size,
                              hipStream_t stream)
{
    const int D = 1024, F = 4096, H = 16, T = 2048;
    const int BT = in_sizes[0] / D;
    const int B  = BT / T;

    const float* x    = (const float*)d_in[0];
    const float* wq   = (const float*)d_in[1];
    const float* bq   = (const float*)d_in[2];
    const float* wk   = (const float*)d_in[3];
    const float* bk   = (const float*)d_in[4];
    const float* wv   = (const float*)d_in[5];
    const float* bv   = (const float*)d_in[6];
    const float* wo   = (const float*)d_in[7];
    const float* bo   = (const float*)d_in[8];
    const float* ln1g = (const float*)d_in[9];
    const float* ln1b = (const float*)d_in[10];
    const float* w1   = (const float*)d_in[11];
    const float* b1   = (const float*)d_in[12];
    const float* w2   = (const float*)d_in[13];
    const float* b2   = (const float*)d_in[14];
    const float* ln2g = (const float*)d_in[15];
    const float* ln2b = (const float*)d_in[16];
    float* out = (float*)d_out;

    // --- deterministic workspace carve-up (bytes) ---
    char* wp = (char*)d_ws;
    auto carve = [&](size_t bytes) {
        char* p = wp; wp += (bytes + 255) & ~(size_t)255; return p;
    };
    const size_t eD = (size_t)BT * D, eF = (size_t)BT * F;
    __bf16* xb   = (__bf16*)carve(eD * 2);
    __bf16* wqT  = (__bf16*)carve((size_t)D * D * 2);
    __bf16* wkT  = (__bf16*)carve((size_t)D * D * 2);
    __bf16* wvT  = (__bf16*)carve((size_t)D * D * 2);
    __bf16* woT  = (__bf16*)carve((size_t)D * D * 2);
    __bf16* w1T  = (__bf16*)carve((size_t)D * F * 2);
    __bf16* w2T  = (__bf16*)carve((size_t)F * D * 2);
    __bf16* Qb   = (__bf16*)carve(eD * 2);
    __bf16* Kb   = (__bf16*)carve(eD * 2);
    __bf16* Vb   = (__bf16*)carve(eD * 2);
    __bf16* ATTb = (__bf16*)carve(eD * 2);
    __bf16* Hb   = (__bf16*)carve(eD * 2);
    __bf16* FF1b = (__bf16*)carve(eF * 2);
    float*  O1f  = (float*)carve(eD * 4);
    float*  Hf   = (float*)carve(eD * 4);
    float*  FF2f = (float*)carve(eD * 4);

    // --- one-time bf16 conversions / weight transposes ---
    cvt_bf16_kernel<<<(int)(eD / 1024), 256, 0, stream>>>(x, xb, (int)eD);
    transpose_bf16_kernel<<<dim3(D / 32, D / 32), 256, 0, stream>>>(wq, wqT, D, D);
    transpose_bf16_kernel<<<dim3(D / 32, D / 32), 256, 0, stream>>>(wk, wkT, D, D);
    transpose_bf16_kernel<<<dim3(D / 32, D / 32), 256, 0, stream>>>(wv, wvT, D, D);
    transpose_bf16_kernel<<<dim3(D / 32, D / 32), 256, 0, stream>>>(wo, woT, D, D);
    transpose_bf16_kernel<<<dim3(F / 32, D / 32), 256, 0, stream>>>(w1, w1T, D, F);
    transpose_bf16_kernel<<<dim3(D / 32, F / 32), 256, 0, stream>>>(w2, w2T, F, D);

    const dim3 gD(D / BN, BT / BM);
    const dim3 gF(F / BN, BT / BM);

    // QKV projections (bf16 out only)
    gemm_bf16_kernel<<<gD, 256, 0, stream>>>(xb, wqT, bq, nullptr, Qb, BT, D, D, 0);
    gemm_bf16_kernel<<<gD, 256, 0, stream>>>(xb, wkT, bk, nullptr, Kb, BT, D, D, 0);
    gemm_bf16_kernel<<<gD, 256, 0, stream>>>(xb, wvT, bv, nullptr, Vb, BT, D, D, 0);
    // causal attention (bf16 in/out)
    attn_kernel<<<dim3(B * H, T / 64), 128, 0, stream>>>(Qb, Kb, Vb, ATTb, B, T, D, H);
    // output projection (f32 out) + LN1 (f32 + bf16 out)
    gemm_bf16_kernel<<<gD, 256, 0, stream>>>(ATTb, woT, bo, O1f, nullptr, BT, D, D, 0);
    ln_kernel<<<BT, 256, 0, stream>>>(x, O1f, ln1g, ln1b, Hf, Hb, D);
    // FFN
    gemm_bf16_kernel<<<gF, 256, 0, stream>>>(Hb,   w1T, b1, nullptr, FF1b, BT, F, D, 1);
    gemm_bf16_kernel<<<gD, 256, 0, stream>>>(FF1b, w2T, b2, FF2f, nullptr, BT, D, F, 0);
    ln_kernel<<<BT, 256, 0, stream>>>(Hf, FF2f, ln2g, ln2b, out, nullptr, D);
}
